// MultiHeadAttention_22557168239121
// MI455X (gfx1250) — matrix operand order repaired and checked by reading, not compiled, not run on hardware
//
#include <hip/hip_runtime.h>
#include <hip/hip_bf16.h>

// ---------------------------------------------------------------------------
// MHA forward for MI455X (gfx1250, wave32, WMMA).
//   DIM=2048, HEADS=16, HEAD_DIM=128, S=4096, B=1.
// Compute-bound (~0.27 TFLOP vs ~200MB traffic): everything on
// v_wmma_f32_16x16x32_bf16.  All GEMMs arranged NT (K contiguous for both
// operands) -> contiguous b128 fragment loads, no transpose loads needed.
// Whole bf16 working set (~110MB) fits in the 192MB L2.
// ---------------------------------------------------------------------------

#define S_LEN 4096
#define DIM_  2048
#define HEADS_ 16
#define HD_   128

typedef __attribute__((ext_vector_type(16))) __bf16 v16bf;
typedef __attribute__((ext_vector_type(8)))  float  v8f;

union FragU { v16bf v; uint4 q[2]; };

__device__ __forceinline__ unsigned short f2bf(float f) {
    unsigned u = __float_as_uint(f);
    unsigned r = u + 0x7FFFu + ((u >> 16) & 1u);   // round-to-nearest-even
    return (unsigned short)(r >> 16);
}

__device__ __forceinline__ v8f zero8() {
    v8f z = {0.f,0.f,0.f,0.f,0.f,0.f,0.f,0.f};
    return z;
}

// A fragment (16x32 bf16, MxK): lane l holds row M=l&15.
// lanes 0-15: elems 0..7 = K 0..7, elems 8..15 = K 16..23
// lanes 16-31: elems 0..7 = K 8..15, elems 8..15 = K 24..31
__device__ __forceinline__ v16bf load_a_frag(const unsigned short* __restrict__ base,
                                             int ld, int row, int kbase, int hi) {
    FragU f;
    const unsigned short* p = base + (size_t)row * ld + kbase + hi * 8;
    f.q[0] = *(const uint4*)(p);
    f.q[1] = *(const uint4*)(p + 16);
    return f.v;
}

// B fragment (32x16 bf16, KxN) from an N-major (ld=K-stride) matrix:
// lane l holds column N=l&15; same K order as load_a_frag (lanes 0-15 K 0..7,16..23; lanes 16-31 K 8..15,24..31).
__device__ __forceinline__ v16bf load_b_frag(const unsigned short* __restrict__ base,
                                             int ld, int nrow, int kbase, int hi) {
    FragU f;
    const unsigned short* p = base + (size_t)nrow * ld + kbase + hi * 8;
    f.q[0] = *(const uint4*)(p);
    f.q[1] = *(const uint4*)(p + 16);
    return f.v;
}

__device__ __forceinline__ v8f wmma_bf16(v16bf a, v16bf b, v8f c) {
    return __builtin_amdgcn_wmma_f32_16x16x32_bf16(
        /*neg_a=*/false, a, /*neg_b=*/false, b,
        /*c_mod=*/(short)0, c, /*reuse_a=*/false, /*reuse_b=*/false);
}

// --------------------------- fp32 -> bf16 convert ---------------------------
__global__ void __launch_bounds__(256)
cvt_f32_bf16_kernel(const float* __restrict__ in, unsigned short* __restrict__ out, int n) {
    int i = blockIdx.x * 256 + threadIdx.x;
    if (i < n) out[i] = f2bf(in[i]);
}

// --------------------------- NT GEMM: Y = A * B^T + bias --------------------
// A: [M,K] bf16 row-major, B: [N,K] bf16 row-major (i.e. torch Linear weight).
// One wave computes a 32(M) x 64(N) tile; 8 accumulators, 8 WMMAs / k-step.
// MODE 0: bf16 row-major out [M,N]; MODE 1: bf16 transposed out [N,M];
// MODE 2: fp32 row-major out [M,N].
template <int MODE>
__global__ void __launch_bounds__(32)
gemm_nt_kernel(const unsigned short* __restrict__ A,
               const unsigned short* __restrict__ B,
               const float* __restrict__ bias,
               void* __restrict__ outp,
               int M, int N, int Kdim) {
    const int lane = threadIdx.x & 31;
    const int hi = lane >> 4;
    const int ln = lane & 15;
    const int mbase = blockIdx.x * 32;
    const int nbase = blockIdx.y * 64;

    v8f acc[2][4];
#pragma unroll
    for (int i = 0; i < 2; ++i)
#pragma unroll
        for (int j = 0; j < 4; ++j) acc[i][j] = zero8();

    for (int k = 0; k < Kdim; k += 32) {
        v16bf a0 = load_a_frag(A, Kdim, mbase + ln,      k, hi);
        v16bf a1 = load_a_frag(A, Kdim, mbase + 16 + ln, k, hi);
#pragma unroll
        for (int j = 0; j < 4; ++j) {
            v16bf bf = load_b_frag(B, Kdim, nbase + j * 16 + ln, k, hi);
            acc[0][j] = wmma_bf16(a0, bf, acc[0][j]);
            acc[1][j] = wmma_bf16(a1, bf, acc[1][j]);
        }
    }

#pragma unroll
    for (int j = 0; j < 4; ++j) {
        const int col = nbase + j * 16 + ln;
        const float bv = bias[col];
#pragma unroll
        for (int i = 0; i < 2; ++i) {
#pragma unroll
            for (int r = 0; r < 8; ++r) {
                const int row = mbase + i * 16 + r + 8 * hi;
                const float val = acc[i][j][r] + bv;
                if (MODE == 0)
                    ((unsigned short*)outp)[(size_t)row * N + col] = f2bf(val);
                else if (MODE == 1)
                    ((unsigned short*)outp)[(size_t)col * M + row] = f2bf(val);
                else
                    ((float*)outp)[(size_t)row * N + col] = val;
            }
        }
    }
}

// --------------------------- flash attention --------------------------------
// One wave per (head, 16-query tile). Streams 32-key blocks:
//   S(16x32) = Q(16x128) * K^T  (8 WMMAs), online softmax,
//   O(16x128) += P(16x32) * V   (8 WMMAs, V pre-transposed so NT).
// Causal masking only on the (uniform) diagonal block; exp() is a TRANS op
// and co-executes with the XDL WMMA pipe.
__global__ void __launch_bounds__(32)
flash_attn_kernel(const unsigned short* __restrict__ Q,
                  const unsigned short* __restrict__ K,
                  const unsigned short* __restrict__ Vt,   // [DIM][S]
                  unsigned short* __restrict__ ctx,        // [S][DIM]
                  const int* __restrict__ is_causal_p) {
    __shared__ __align__(16) unsigned short Pld[16 * 32];

    const int lane  = threadIdx.x & 31;
    const int hi    = lane >> 4;
    const int ln    = lane & 15;
    const int qbase = blockIdx.x * 16;
    const int h     = blockIdx.y;
    const int hoff  = h * HD_;
    const int causal = *is_causal_p;
    const float scale = 0.08838834764831845f;   // 1/sqrt(128)

    // Q fragments for this 16-row tile (16 x 128 = 4 K-chunks)
    v16bf qf[4];
#pragma unroll
    for (int kc = 0; kc < 4; ++kc)
        qf[kc] = load_a_frag(Q, DIM_, qbase + ln, hoff + kc * 32, hi);

    v8f O[8];
#pragma unroll
    for (int t = 0; t < 8; ++t) O[t] = zero8();
    float rowmax[8], rowsum[8];
#pragma unroll
    for (int r = 0; r < 8; ++r) { rowmax[r] = -1e30f; rowsum[r] = 0.f; }

    const int kend = causal ? (qbase + 16) : S_LEN;

    for (int k0 = 0; k0 < kend; k0 += 32) {
        // prefetch next key/value block a step ahead (gfx1250 global_prefetch_b8;
        // no LOADcnt cost, warms WGP$/L2)
        if (k0 + 32 < kend) {
            __builtin_prefetch(K  + (size_t)(k0 + 32 + ln) * DIM_ + hoff, 0, 1);
            __builtin_prefetch(Vt + (size_t)(hoff + ln) * S_LEN + k0 + 32, 0, 1);
        }

        // ---- S tile = Q * K^T ----
        v8f s0 = zero8(), s1 = zero8();
#pragma unroll
        for (int kc = 0; kc < 4; ++kc) {
            v16bf kf0 = load_b_frag(K, DIM_, k0 + ln,      hoff + kc * 32, hi);
            v16bf kf1 = load_b_frag(K, DIM_, k0 + 16 + ln, hoff + kc * 32, hi);
            s0 = wmma_bf16(qf[kc], kf0, s0);
            s1 = wmma_bf16(qf[kc], kf1, s1);
        }

        // mask is needed only when this block straddles the diagonal — this is
        // a wave-uniform condition, so it becomes a scalar branch.
        const bool diag = causal && (k0 + 16 > qbase);

        // ---- online softmax over this 32-key block ----
        float p0[8], p1[8];
#pragma unroll
        for (int r = 0; r < 8; ++r) {
            const int qrow = qbase + r + 8 * hi;            // C-layout row
            float a = s0[r] * scale;
            float b = s1[r] * scale;
            if (diag) {
                if (k0 + ln      > qrow) a = -1e30f;
                if (k0 + 16 + ln > qrow) b = -1e30f;
            }
            float mx = fmaxf(a, b);
#pragma unroll
            for (int off = 1; off < 16; off <<= 1)
                mx = fmaxf(mx, __shfl_xor(mx, off, 32));
            const float mnew = fmaxf(rowmax[r], mx);
            const float corr = __expf(rowmax[r] - mnew);
            const float e0 = __expf(a - mnew);
            const float e1 = __expf(b - mnew);
            float ps = e0 + e1;
#pragma unroll
            for (int off = 1; off < 16; off <<= 1)
                ps += __shfl_xor(ps, off, 32);
            rowsum[r] = rowsum[r] * corr + ps;
            rowmax[r] = mnew;
            p0[r] = e0; p1[r] = e1;
#pragma unroll
            for (int t = 0; t < 8; ++t) O[t][r] *= corr;    // rescale out acc
        }

        // ---- C-layout P -> LDS -> A-layout fragment ----
#pragma unroll
        for (int r = 0; r < 8; ++r) {
            const int row = r + 8 * hi;
            Pld[row * 32 + ln]      = f2bf(p0[r]);
            Pld[row * 32 + 16 + ln] = f2bf(p1[r]);
        }
        __syncthreads();
        v16bf pf = load_a_frag((const unsigned short*)Pld, 32, ln, 0, hi);

        // ---- O += P * V (Vt rows are d, contiguous over keys -> NT) ----
#pragma unroll
        for (int t = 0; t < 8; ++t) {
            v16bf vf = load_b_frag(Vt, S_LEN, hoff + t * 16 + ln, k0, hi);
            O[t] = wmma_bf16(pf, vf, O[t]);
        }
        __syncthreads();
    }

    // ---- normalize + store ctx (bf16 [S][DIM]) ----
#pragma unroll
    for (int t = 0; t < 8; ++t) {
#pragma unroll
        for (int r = 0; r < 8; ++r) {
            const int row = qbase + r + 8 * hi;
            const float val = O[t][r] / rowsum[r];
            ctx[(size_t)row * DIM_ + hoff + t * 16 + ln] = f2bf(val);
        }
    }
}

// --------------------------- launch -----------------------------------------
extern "C" void kernel_launch(void* const* d_in, const int* in_sizes, int n_in,
                              void* d_out, int out_size, void* d_ws, size_t ws_size,
                              hipStream_t stream) {
    const float* X  = (const float*)d_in[0];
    const float* Wq = (const float*)d_in[1];
    const float* bq = (const float*)d_in[2];
    const float* Wk = (const float*)d_in[3];
    const float* bk = (const float*)d_in[4];
    const float* Wv = (const float*)d_in[5];
    const float* bv = (const float*)d_in[6];
    const float* Wo = (const float*)d_in[7];
    const float* bo = (const float*)d_in[8];
    const int* is_causal = (const int*)d_in[9];

    const int SD = S_LEN * DIM_;       // 8M elems
    const int DD = DIM_ * DIM_;        // 4M elems

    char* ws = (char*)d_ws;
    size_t off = 0;
    auto alloc = [&](size_t bytes) -> void* {
        void* p = ws + off;
        off += (bytes + 255) & ~(size_t)255;
        return p;
    };
    unsigned short* Xb  = (unsigned short*)alloc((size_t)SD * 2);
    unsigned short* Wqb = (unsigned short*)alloc((size_t)DD * 2);
    unsigned short* Wkb = (unsigned short*)alloc((size_t)DD * 2);
    unsigned short* Wvb = (unsigned short*)alloc((size_t)DD * 2);
    unsigned short* Wob = (unsigned short*)alloc((size_t)DD * 2);
    unsigned short* Qb  = (unsigned short*)alloc((size_t)SD * 2);
    unsigned short* Kb  = (unsigned short*)alloc((size_t)SD * 2);
    unsigned short* Vtb = (unsigned short*)alloc((size_t)SD * 2);  // [DIM][S]
    unsigned short* Ctx = (unsigned short*)alloc((size_t)SD * 2);

    // 1) fp32 -> bf16
    cvt_f32_bf16_kernel<<<(SD + 255) / 256, 256, 0, stream>>>(X,  Xb,  SD);
    cvt_f32_bf16_kernel<<<(DD + 255) / 256, 256, 0, stream>>>(Wq, Wqb, DD);
    cvt_f32_bf16_kernel<<<(DD + 255) / 256, 256, 0, stream>>>(Wk, Wkb, DD);
    cvt_f32_bf16_kernel<<<(DD + 255) / 256, 256, 0, stream>>>(Wv, Wvb, DD);
    cvt_f32_bf16_kernel<<<(DD + 255) / 256, 256, 0, stream>>>(Wo, Wob, DD);

    // 2) projections: Q/K row-major, V stored transposed for NT attention GEMM
    dim3 gp(S_LEN / 32, DIM_ / 64);
    gemm_nt_kernel<0><<<gp, 32, 0, stream>>>(Xb, Wqb, bq, Qb,  S_LEN, DIM_, DIM_);
    gemm_nt_kernel<0><<<gp, 32, 0, stream>>>(Xb, Wkb, bk, Kb,  S_LEN, DIM_, DIM_);
    gemm_nt_kernel<1><<<gp, 32, 0, stream>>>(Xb, Wvb, bv, Vtb, S_LEN, DIM_, DIM_);

    // 3) flash attention: one wave per (16-query tile, head)
    flash_attn_kernel<<<dim3(S_LEN / 16, HEADS_), 32, 0, stream>>>(
        Qb, Kb, Vtb, Ctx, is_causal);

    // 4) output projection, fp32 result
    gemm_nt_kernel<2><<<gp, 32, 0, stream>>>(Ctx, Wob, bo, d_out, S_LEN, DIM_, DIM_);
}
